// CentroidAttention_3891240370359
// MI455X (gfx1250) — matrix operand order repaired and checked by reading, not compiled, not run on hardware
//
#include <hip/hip_runtime.h>

// ---------------------------------------------------------------------------
// CDNA5 (gfx1250) CentroidAttention: bf16 WMMA pipeline, wave32,
// async global->LDS staging of the B operand.
// ---------------------------------------------------------------------------

typedef unsigned short u16;
typedef __attribute__((ext_vector_type(16))) __bf16 v16bf;
typedef __attribute__((ext_vector_type(8)))  float  v8f;

union FragU { uint4 u[2]; v16bf v; };

static constexpr int BATCH   = 16384;
static constexpr int FDIM    = 1024;
static constexpr int ADIM    = 512;
static constexpr int CVALID  = 1000;
static constexpr int CPAD    = 1024;

// LDS B-tile: 128 columns x 64 K, padded row stride 72 u16 (144 B, 16B-aligned,
// 36-bank step -> conflict-free across 16 lanes for b128 reads).
static constexpr int BSTRIDE = 72;

__device__ __forceinline__ u16 f2bf(float f) {
  unsigned u = __float_as_uint(f);
  u += 0x7fffu + ((u >> 16) & 1u);          // round-to-nearest-even
  return (u16)(u >> 16);
}

// ---------------------------------------------------------------------------
// 1) features -> d_out[:, 0:1024] (f32)  and  features -> bf16 copy
// ---------------------------------------------------------------------------
__global__ __launch_bounds__(256) void feat_pre(const float* __restrict__ f,
                                                float* __restrict__ out,
                                                u16* __restrict__ fb) {
  size_t idx = (size_t)blockIdx.x * 256 + threadIdx.x;   // BATCH*256 float4 units
  size_t b = idx >> 8; int j = (int)(idx & 255);
  float4 x = ((const float4*)f)[idx];
  ((float4*)out)[b * 512 + j] = x;                       // ldc = 2048 f32 = 512 float4
  uint2 p;
  p.x = (unsigned)f2bf(x.x) | ((unsigned)f2bf(x.y) << 16);
  p.y = (unsigned)f2bf(x.z) | ((unsigned)f2bf(x.w) << 16);
  ((uint2*)fb)[idx] = p;
}

// ---------------------------------------------------------------------------
// 2) transpose + convert weight: in [R,C] f32 -> out [C,R] bf16 (B operand layout)
// ---------------------------------------------------------------------------
__global__ __launch_bounds__(256) void transconv(const float* __restrict__ in,
                                                 u16* __restrict__ out,
                                                 int R, int C) {
  int idx = blockIdx.x * 256 + threadIdx.x;              // C*R total
  int n = idx / R, k = idx % R;
  out[idx] = f2bf(in[(size_t)k * C + n]);
}

// ---------------------------------------------------------------------------
// 3) class centers: counts + feature sums (f32 atomics), then finalize to bf16
// ---------------------------------------------------------------------------
__global__ __launch_bounds__(256) void centers_count(const int* __restrict__ labels,
                                                     float* __restrict__ counts) {
  int b = blockIdx.x * 256 + threadIdx.x;
  atomicAdd(&counts[labels[b]], 1.0f);
}

__global__ __launch_bounds__(256) void centers_accum(const float* __restrict__ f,
                                                     const int* __restrict__ labels,
                                                     float* __restrict__ sums) {
  size_t idx = (size_t)blockIdx.x * 256 + threadIdx.x;   // BATCH*256 float4 units
  size_t b = idx >> 8; int j = (int)(idx & 255);
  int lab = labels[b];
  float4 x = ((const float4*)f)[idx];
  float* s = sums + (size_t)lab * FDIM + j * 4;
  atomicAdd(s + 0, x.x); atomicAdd(s + 1, x.y);
  atomicAdd(s + 2, x.z); atomicAdd(s + 3, x.w);
}

__global__ __launch_bounds__(256) void centers_final(const float* __restrict__ sums,
                                                     const float* __restrict__ counts,
                                                     u16* __restrict__ cbf) {
  int idx = blockIdx.x * 256 + threadIdx.x;              // CPAD*FDIM
  int c = idx >> 10;
  float cnt = (c < CVALID) ? counts[c] : 0.0f;
  float v = (cnt > 0.0f) ? sums[idx] * (1.0f / cnt) : 0.0f;
  cbf[idx] = f2bf(v);
}

// ---------------------------------------------------------------------------
// 4) bf16 WMMA GEMM with async-LDS B staging:
//      C[M,N] = A[M,K] @ B[K,N], B given as Bt[N,K] (row = B column, K-contig).
//    Block = 256 threads = 8 waves; wave tile 32x64; block tile 128x128.
//    Per 64-K step: B tile (128x64 bf16) staged via GLOBAL_LOAD_ASYNC_TO_LDS_B128
//    (ASYNCcnt) and shared by all waves; A fragments stream from global.
//    OUT_MODE: 0 = f32 store, 1 = bf16 store, 2 = bf16 transposed store (C^T).
// ---------------------------------------------------------------------------
template <int OUT_MODE, bool ADD_BIAS>
__global__ __launch_bounds__(256) void gemm_bf16_wmma(
    const u16* __restrict__ A, int lda,
    const u16* __restrict__ Bt, int ldb,
    void* __restrict__ Cout, int ldc, int col0,
    const float* __restrict__ bias, int K) {
  __shared__ u16 tile[128 * BSTRIDE];

  const int t    = threadIdx.x;
  const int lane = t & 31;
  const int wid  = t >> 5;
  const int half = lane >> 4;        // 0: lanes 0-15, 1: lanes 16-31
  const int r    = lane & 15;

  const int bn = blockIdx.x * 128;
  const int wm = blockIdx.y * 128 + (wid & 3) * 32;   // wave M strip (32 rows)
  const int gN = (wid >> 2) * 64;                     // wave N group within block

  // A fragment (16x32 bf16): lane holds row r, K runs [half*8,+8) and [16+half*8,+8)
  const u16* Ar0 = A + (size_t)(wm + r) * lda + half * 8;
  const u16* Ar1 = Ar0 + (size_t)16 * lda;

  v8f acc[2][4];
#pragma unroll
  for (int tm = 0; tm < 2; ++tm)
#pragma unroll
    for (int tn = 0; tn < 4; ++tn) acc[tm][tn] = (v8f){};

  for (int k0 = 0; k0 < K; k0 += 64) {
    __syncthreads();   // previous tile fully consumed before overwrite

    // ---- stage B tile: 128 rows(cols of B) x 64 K, 16 KB, 4 x b128 per thread
#pragma unroll
    for (int i = 0; i < 4; ++i) {
      const int c   = t + i * 256;       // chunk id, 1024 x 16B
      const int row = c >> 3;            // 8 chunks per 64-u16 row
      const int ko  = (c & 7) * 8;       // u16 offset within row
      unsigned laddr = (unsigned)(size_t)&tile[row * BSTRIDE + ko];
      const u16* gp  = Bt + (size_t)(bn + row) * ldb + k0 + ko;
      asm volatile("global_load_async_to_lds_b128 %0, %1, off"
                   :: "v"(laddr), "v"(gp) : "memory");
    }
    asm volatile("s_wait_asynccnt 0x0" ::: "memory");
    __syncthreads();

    // speculative prefetch of next A tile (OOB-safe by spec)
    __builtin_prefetch(Ar0 + k0 + 64, 0, 0);
    __builtin_prefetch(Ar1 + k0 + 64, 0, 0);

    // ---- compute: two 32-K substeps, 8 WMMA each
#pragma unroll
    for (int ks = 0; ks < 2; ++ks) {
      const int kk = k0 + ks * 32;
      FragU a0, a1;
      a0.u[0] = *(const uint4*)(Ar0 + kk);
      a0.u[1] = *(const uint4*)(Ar0 + kk + 16);
      a1.u[0] = *(const uint4*)(Ar1 + kk);
      a1.u[1] = *(const uint4*)(Ar1 + kk + 16);
#pragma unroll
      for (int tn = 0; tn < 4; ++tn) {
        // B fragment (32x16): lane holds column r of sub-tile, K runs [half*8,+8) and [16+half*8,+8)
        const u16* bp = &tile[(gN + tn * 16 + r) * BSTRIDE + ks * 32 + half * 8];
        FragU b;
        b.u[0] = *(const uint4*)bp;
        b.u[1] = *(const uint4*)(bp + 16);
        acc[0][tn] = __builtin_amdgcn_wmma_f32_16x16x32_bf16(false, a0.v, false, b.v, (short)0, acc[0][tn], false, false);
        acc[1][tn] = __builtin_amdgcn_wmma_f32_16x16x32_bf16(false, a1.v, false, b.v, (short)0, acc[1][tn], false, false);
      }
    }
  }

#pragma unroll
  for (int tm = 0; tm < 2; ++tm) {
#pragma unroll
    for (int tn = 0; tn < 4; ++tn) {
      const int mbase = wm + tm * 16 + half * 8;     // C layout: VGPR j -> M = mbase+j
      const int n     = bn + gN + tn * 16 + r;       // lane -> N
      const float bv  = ADD_BIAS ? bias[n] : 0.0f;
#pragma unroll
      for (int j = 0; j < 8; ++j) {
        float vo = acc[tm][tn][j] + bv;
        if (OUT_MODE == 0)
          ((float*)Cout)[(size_t)(mbase + j) * ldc + col0 + n] = vo;
        else if (OUT_MODE == 1)
          ((u16*)Cout)[(size_t)(mbase + j) * ldc + col0 + n] = f2bf(vo);
        else
          ((u16*)Cout)[(size_t)n * ldc + mbase + j] = f2bf(vo);
      }
    }
  }
}

// ---------------------------------------------------------------------------
// 5) row softmax, in place: read f32 logits row, write bf16 probs into same row.
//    Safe: all reads happen before the first barrier; writes after the last.
// ---------------------------------------------------------------------------
__global__ __launch_bounds__(256) void softmax_inplace(float* __restrict__ logits,
                                                       float scale) {
  __shared__ float red[8];
  const int t = threadIdx.x;
  const int lane = t & 31, wid = t >> 5;
  float* rp = logits + (size_t)blockIdx.x * CPAD;

  float4 x = ((const float4*)rp)[t];
  float xa[4] = {x.x, x.y, x.z, x.w};
  float v[4];
  float lm = -1e30f;
#pragma unroll
  for (int i = 0; i < 4; ++i) {
    int c = t * 4 + i;
    v[i] = (c < CVALID) ? xa[i] * scale : -1e30f;
    lm = fmaxf(lm, v[i]);
  }
#pragma unroll
  for (int off = 16; off > 0; off >>= 1) lm = fmaxf(lm, __shfl_xor(lm, off, 32));
  if (lane == 0) red[wid] = lm;
  __syncthreads();
  if (t == 0) {
    float m = red[0];
#pragma unroll
    for (int i = 1; i < 8; ++i) m = fmaxf(m, red[i]);
    red[0] = m;
  }
  __syncthreads();
  const float mx = red[0];
  __syncthreads();

  float e[4]; float ls = 0.0f;
#pragma unroll
  for (int i = 0; i < 4; ++i) { e[i] = __expf(v[i] - mx); ls += e[i]; }
#pragma unroll
  for (int off = 16; off > 0; off >>= 1) ls += __shfl_xor(ls, off, 32);
  if (lane == 0) red[wid] = ls;
  __syncthreads();
  if (t == 0) {
    float s = 0.0f;
#pragma unroll
    for (int i = 0; i < 8; ++i) s += red[i];
    red[0] = s;
  }
  __syncthreads();
  const float inv = 1.0f / red[0];

  uint2 p;
  p.x = (unsigned)f2bf(e[0] * inv) | ((unsigned)f2bf(e[1] * inv) << 16);
  p.y = (unsigned)f2bf(e[2] * inv) | ((unsigned)f2bf(e[3] * inv) << 16);
  ((uint2*)rp)[t] = p;   // bf16 P packed into first half of each f32 row
}

// ---------------------------------------------------------------------------
// launcher
// ---------------------------------------------------------------------------
extern "C" void kernel_launch(void* const* d_in, const int* in_sizes, int n_in,
                              void* d_out, int out_size, void* d_ws, size_t ws_size,
                              hipStream_t stream) {
  (void)in_sizes; (void)n_in; (void)out_size; (void)ws_size;
  const float* features = (const float*)d_in[0];
  const int*   labels   = (const int*)d_in[1];
  const float* Wq       = (const float*)d_in[2];
  const float* Wk       = (const float*)d_in[3];
  const float* Wv       = (const float*)d_in[4];
  const float* Wproj    = (const float*)d_in[5];
  const float* bproj    = (const float*)d_in[6];
  float* out = (float*)d_out;

  char* ws = (char*)d_ws;
  size_t off = 0;
  auto alloc = [&](size_t bytes) { void* p = ws + off; off = (off + bytes + 255) & ~(size_t)255; return p; };

  float* sums   = (float*)alloc((size_t)CPAD * FDIM * 4);
  float* counts = (float*)alloc((size_t)CPAD * 4);
  u16*   cbf    = (u16*)  alloc((size_t)CPAD * FDIM * 2);   // centers bf16 [1024,1024]
  u16*   fbf    = (u16*)  alloc((size_t)BATCH * FDIM * 2);  // features bf16
  u16*   WqT    = (u16*)  alloc((size_t)ADIM * FDIM * 2);   // [512,1024]
  u16*   WkT    = (u16*)  alloc((size_t)ADIM * FDIM * 2);
  u16*   WvT    = (u16*)  alloc((size_t)ADIM * FDIM * 2);
  u16*   WpT    = (u16*)  alloc((size_t)FDIM * ADIM * 2);   // [1024,512]
  u16*   qb     = (u16*)  alloc((size_t)BATCH * ADIM * 2);  // q bf16; reused for attn
  u16*   kb     = (u16*)  alloc((size_t)CPAD * ADIM * 2);   // k bf16 [1024,512]
  u16*   vT     = (u16*)  alloc((size_t)ADIM * CPAD * 2);   // v^T bf16 [512,1024]
  float* logits = (float*)alloc((size_t)BATCH * CPAD * 4);  // also holds bf16 P in place

  // features -> out[:, :1024] + bf16
  feat_pre<<<BATCH, 256, 0, stream>>>(features, out, fbf);
  // weights -> transposed bf16
  transconv<<<(ADIM * FDIM) / 256, 256, 0, stream>>>(Wq, WqT, FDIM, ADIM);
  transconv<<<(ADIM * FDIM) / 256, 256, 0, stream>>>(Wk, WkT, FDIM, ADIM);
  transconv<<<(ADIM * FDIM) / 256, 256, 0, stream>>>(Wv, WvT, FDIM, ADIM);
  transconv<<<(FDIM * ADIM) / 256, 256, 0, stream>>>(Wproj, WpT, ADIM, FDIM);
  // centers
  hipMemsetAsync(sums, 0, (size_t)CPAD * FDIM * 4 + 4096, stream);  // sums + counts
  centers_count<<<BATCH / 256, 256, 0, stream>>>(labels, counts);
  centers_accum<<<BATCH, 256, 0, stream>>>(features, labels, sums);
  centers_final<<<(CPAD * FDIM) / 256, 256, 0, stream>>>(sums, counts, cbf);

  // k = centers @ Wk          [1024,512] bf16
  gemm_bf16_wmma<1, false><<<dim3(ADIM / 128, CPAD / 128), 256, 0, stream>>>(
      cbf, FDIM, WkT, FDIM, kb, ADIM, 0, nullptr, FDIM);
  // v^T = (centers @ Wv)^T    [512,1024] bf16
  gemm_bf16_wmma<2, false><<<dim3(ADIM / 128, CPAD / 128), 256, 0, stream>>>(
      cbf, FDIM, WvT, FDIM, vT, CPAD, 0, nullptr, FDIM);
  // q = features @ Wq         [16384,512] bf16
  gemm_bf16_wmma<1, false><<<dim3(ADIM / 128, BATCH / 128), 256, 0, stream>>>(
      fbf, FDIM, WqT, FDIM, qb, ADIM, 0, nullptr, FDIM);
  // logits = q @ k^T          [16384,1024] f32 (k rows are B columns)
  gemm_bf16_wmma<0, false><<<dim3(CPAD / 128, BATCH / 128), 256, 0, stream>>>(
      qb, ADIM, kb, ADIM, logits, CPAD, 0, nullptr, ADIM);
  // softmax (scale folded in), P bf16 written in place over logits rows
  softmax_inplace<<<BATCH, 256, 0, stream>>>(logits, 0.044194173824159f /* 512^-0.5 */);
  // attn = P @ v              [16384,512] bf16 (reuse q buffer; P lda = 2048 u16)
  gemm_bf16_wmma<1, false><<<dim3(ADIM / 128, BATCH / 128), 256, 0, stream>>>(
      (const u16*)logits, 2 * CPAD, vT, CPAD, qb, ADIM, 0, nullptr, CPAD);
  // out[:, 1024:2048] = attn @ Wproj + bproj   (f32, ldc = 2048, col offset 1024)
  gemm_bf16_wmma<0, true><<<dim3(FDIM / 128, BATCH / 128), 256, 0, stream>>>(
      qb, ADIM, WpT, ADIM, out, 2 * FDIM, FDIM, bproj, ADIM);
}
